// AttentionModel_27462020891183
// MI455X (gfx1250) — matrix operand order repaired and checked by reading, not compiled, not run on hardware
//
#include <hip/hip_runtime.h>

// ---------------- model constants (match reference) ----------------
constexpr int B  = 4;
constexpr int T  = 512;
constexpr int D  = 1024;
constexpr int H  = 16;
constexpr int L  = 6;
constexpr int V  = 8000;
constexpr int HD = D / H;     // 64
constexpr int FF = 4 * D;     // 4096
constexpr int MT = B * T;     // 2048 rows of activations

typedef __attribute__((ext_vector_type(16))) __bf16 bf16x16;
typedef __attribute__((ext_vector_type(8)))  __bf16 bf16x8;
typedef __attribute__((ext_vector_type(8)))  float  f32x8;
typedef __attribute__((ext_vector_type(4)))  int    i32x4;

#define AS1 __attribute__((address_space(1)))
#define AS3 __attribute__((address_space(3)))

#if defined(__has_builtin)
#if __has_builtin(__builtin_amdgcn_global_load_async_to_lds_b128)
#define HAVE_ASYNC 1
#endif
#endif
#ifndef HAVE_ASYNC
#define HAVE_ASYNC 0
#endif

// ---------------- WMMA helpers ----------------
__device__ __forceinline__ f32x8 wmma_bf16(bf16x16 a, bf16x16 b, f32x8 c) {
  return __builtin_amdgcn_wmma_f32_16x16x32_bf16(
      /*neg_a=*/false, a, /*neg_b=*/false, b,
      /*c_mod=*/(short)0, c, /*reuse_a=*/false, /*reuse_b=*/false);
}

// A-operand fragment (16x32, 16-bit): lane<16 holds K={0..7,16..23}, lane>=16
// holds K={8..15,24..31}. Caller passes p already offset by (row*lda + half*8).
__device__ __forceinline__ bf16x16 ld_frag_a(const __bf16* p) {
  bf16x8 lo = *(const bf16x8*)p;
  bf16x8 hi = *(const bf16x8*)(p + 16);
  bf16x16 r;
#pragma unroll
  for (int i = 0; i < 8; i++) { r[i] = lo[i]; r[i + 8] = hi[i]; }
  return r;
}

// B-operand fragment (32x16, 16-bit, supplied as Bt[N,K]): lane holds column
// n: lanes 0-15 K={0..7,16..23}, lanes 16-31 K={8..15,24..31}. Caller
// passes p offset by (nrow*ldb + half*8).
__device__ __forceinline__ bf16x16 ld_frag_b(const __bf16* p) {
  bf16x8 lo = *(const bf16x8*)p;
  bf16x8 hi = *(const bf16x8*)(p + 16);
  bf16x16 r;
#pragma unroll
  for (int i = 0; i < 8; i++) { r[i] = lo[i]; r[i + 8] = hi[i]; }
  return r;
}

__device__ __forceinline__ float redmax16(float v) {
#pragma unroll
  for (int m = 1; m <= 8; m <<= 1) v = fmaxf(v, __shfl_xor(v, m, 32));
  return v;
}
__device__ __forceinline__ float redsum16(float v) {
#pragma unroll
  for (int m = 1; m <= 8; m <<= 1) v += __shfl_xor(v, m, 32);
  return v;
}

// 16-byte global -> LDS copy: async DMA (ASYNCcnt) when available.
__device__ __forceinline__ void copy16_g2l(const __bf16* g, __bf16* l) {
#if HAVE_ASYNC
  __builtin_amdgcn_global_load_async_to_lds_b128(
      (AS1 i32x4*)(AS1 void*)(void*)g,
      (AS3 i32x4*)(AS3 void*)(void*)l, 0, 0);
#else
  *(bf16x8*)l = *(const bf16x8*)g;
#endif
}
__device__ __forceinline__ void async_wait_le1() {
#if HAVE_ASYNC
  asm volatile("s_wait_asynccnt 0x1" ::: "memory");
#endif
}
__device__ __forceinline__ void async_wait_le0() {
#if HAVE_ASYNC
  asm volatile("s_wait_asynccnt 0x0" ::: "memory");
#endif
}

// ---------------- embedding ----------------
__global__ __launch_bounds__(256)
void embed_ker(const int* __restrict__ tok, const float* __restrict__ te,
               const float* __restrict__ pe, float* __restrict__ x) {
  int row = blockIdx.x;            // b*T + t
  int t   = row & (T - 1);
  int id  = tok[row];
  const float* tr = te + (size_t)id * D;
  const float* pr = pe + (size_t)t * D;
  float* xr = x + (size_t)row * D;
  for (int i = threadIdx.x; i < D; i += 256) xr[i] = tr[i] + pr[i];
}

// ---------------- layernorm (f32 in -> bf16 out) ----------------
__global__ __launch_bounds__(256)
void ln_bf16_ker(const float* __restrict__ x, const float* __restrict__ g,
                 const float* __restrict__ bb, __bf16* __restrict__ out) {
  __shared__ float sred[256];
  int row = blockIdx.x;
  const float* xr = x + (size_t)row * D;
  float s = 0.f, ss = 0.f;
  for (int i = threadIdx.x; i < D; i += 256) { float v = xr[i]; s += v; ss += v * v; }
  sred[threadIdx.x] = s; __syncthreads();
  for (int off = 128; off; off >>= 1) {
    if (threadIdx.x < off) sred[threadIdx.x] += sred[threadIdx.x + off];
    __syncthreads();
  }
  float mean = sred[0] / D; __syncthreads();
  sred[threadIdx.x] = ss; __syncthreads();
  for (int off = 128; off; off >>= 1) {
    if (threadIdx.x < off) sred[threadIdx.x] += sred[threadIdx.x + off];
    __syncthreads();
  }
  float var  = sred[0] / D - mean * mean;
  float rstd = rsqrtf(var + 1e-5f);
  __bf16* orow = out + (size_t)row * D;
  for (int i = threadIdx.x; i < D; i += 256)
    orow[i] = (__bf16)((xr[i] - mean) * rstd * g[i] + bb[i]);
}

// ---------------- transpose + f32->bf16 : Wt[n*K + k] = W[k*N + n] ----------
__global__ __launch_bounds__(256)
void transpose_cvt_ker(const float* __restrict__ W, __bf16* __restrict__ Wt,
                       int K, int N) {
  __shared__ float tile[32][33];
  int nb = blockIdx.x * 32, kb = blockIdx.y * 32;
  int tx = threadIdx.x & 31, ty = threadIdx.x >> 5;  // 32 x 8
#pragma unroll
  for (int j = 0; j < 4; j++)
    tile[ty + 8 * j][tx] = W[(size_t)(kb + ty + 8 * j) * N + nb + tx];
  __syncthreads();
#pragma unroll
  for (int j = 0; j < 4; j++)
    Wt[(size_t)(nb + ty + 8 * j) * K + kb + tx] = (__bf16)tile[tx][ty + 8 * j];
}

// ---------------- generic WMMA GEMM: C[M,N] = A[M,K] @ Bt[N,K]^T ------------
// Block = 256 threads (8 waves), tile 128(M) x 64(N). The shared 64x32 B tile
// is staged into LDS (double-buffered) with async global->LDS DMA; all waves
// read B via ds_load_b128, cutting global B traffic 8x and overlapping the
// next stage's DMA with the current stage's WMMAs.
constexpr int MB_BIAS = 1, MB_RES = 2, MB_RELU = 4, MB_BF16 = 8, MB_VT = 16;
constexpr int ROWB = 80;   // 64B of K-data + 16B pad (20-dword bank stride)

template <int MODE>
__global__ __launch_bounds__(256)
void gemm_bf16_ker(const __bf16* __restrict__ A, const __bf16* __restrict__ Bt,
                   const float* __restrict__ bias, const float* __restrict__ R,
                   float* __restrict__ Cf, __bf16* __restrict__ Cb,
                   int M, int N, int K) {
  __shared__ __align__(16) char bsm[2 * 64 * ROWB];
  int tid  = threadIdx.x;
  int lane = tid & 31;
  int wid  = tid >> 5;        // 0..7
  int half = lane >> 4;
  int l16  = lane & 15;
  int m0 = blockIdx.y * 128 + wid * 16;
  int n0 = blockIdx.x * 64;

  f32x8 cc[4];
#pragma unroll
  for (int i = 0; i < 4; i++)
#pragma unroll
    for (int r = 0; r < 8; r++) cc[i][r] = 0.f;

  const __bf16* Ap = A + (size_t)(m0 + l16) * K + half * 8;

  // staging assignment: thread -> (row, 16B chunk) of the 64x64B tile
  int srow = tid >> 2;        // 0..63
  int schk = tid & 3;         // 0..3
  const __bf16* gsrc = Bt + (size_t)(n0 + srow) * K + schk * 8;
  char* ldst = bsm + srow * ROWB + schk * 16;

  int nk = K / 32;
  copy16_g2l(gsrc, (__bf16*)ldst);                       // stage 0 -> buf 0
  for (int ks = 0; ks < nk; ++ks) {
    if (ks + 1 < nk) {
      copy16_g2l(gsrc + (size_t)(ks + 1) * 32,
                 (__bf16*)(ldst + ((ks + 1) & 1) * 64 * ROWB));
      async_wait_le1();       // stage ks done (in-order), stage ks+1 in flight
    } else {
      async_wait_le0();
    }
    __syncthreads();          // all waves' stage-ks chunks visible

    const char* buf = bsm + (ks & 1) * 64 * ROWB;
    bf16x16 a = ld_frag_a(Ap + (size_t)ks * 32);
#pragma unroll
    for (int nf = 0; nf < 4; nf++) {
      const __bf16* bp =
          (const __bf16*)(buf + (nf * 16 + l16) * ROWB + half * 16);
      bf16x16 b = ld_frag_b(bp);
      cc[nf] = wmma_bf16(a, b, cc[nf]);
    }
    __syncthreads();          // done reading before buffer is re-staged
  }

#pragma unroll
  for (int nf = 0; nf < 4; nf++) {
#pragma unroll
    for (int r = 0; r < 8; r++) {
      int m = m0 + r + 8 * half;
      int n = n0 + nf * 16 + l16;
      float v = cc[nf][r];
      if (MODE & MB_BIAS) v += bias[n];
      if (MODE & MB_RELU) v = fmaxf(v, 0.f);
      if (MODE & MB_RES)  v += R[(size_t)m * N + n];
      if (MODE & MB_VT) {
        // write V transposed per head: vt[b][h][d][t], T=512, H=16, HD=64
        int bb2 = m >> 9, tt = m & 511, hh = n >> 6, dd = n & 63;
        Cb[(((size_t)(bb2 * H + hh)) * HD + dd) * T + tt] = (__bf16)v;
      } else if (MODE & MB_BF16) {
        Cb[(size_t)m * N + n] = (__bf16)v;
      } else {
        Cf[(size_t)m * N + n] = v;
      }
    }
  }
}

// ---------------- flash attention: one wave per (b, h, 16-query block) -----
__global__ __launch_bounds__(32)
void attn_ker(const __bf16* __restrict__ q, const __bf16* __restrict__ k,
              const __bf16* __restrict__ vt, __bf16* __restrict__ y) {
  __shared__ __bf16 pbuf[16 * 32];
  int lane = threadIdx.x & 31;
  int half = lane >> 4, l16 = lane & 15;
  int qbi = blockIdx.x & 31;           // T/16 = 32
  int h   = (blockIdx.x >> 5) & 15;
  int b   = blockIdx.x >> 9;
  int qb  = qbi * 16;

  size_t qrow = (size_t)(b * T + qb + l16) * D + h * HD;
  bf16x16 a0 = ld_frag_a(q + qrow + half * 8);        // dims 0..31
  bf16x16 a1 = ld_frag_a(q + qrow + 32 + half * 8);   // dims 32..63

  f32x8 o[4];
#pragma unroll
  for (int i = 0; i < 4; i++)
#pragma unroll
    for (int r = 0; r < 8; r++) o[i][r] = 0.f;

  float rmax[8], rsum[8];
#pragma unroll
  for (int r = 0; r < 8; r++) { rmax[r] = -__builtin_inff(); rsum[r] = 0.f; }

  const __bf16* vbase = vt + (size_t)(b * H + h) * HD * T;

  for (int kb = 0; kb < qb + 16; kb += 32) {
    // ---- S = Q K^T for two 16-key subtiles ----
    f32x8 s0, s1;
#pragma unroll
    for (int r = 0; r < 8; r++) { s0[r] = 0.f; s1[r] = 0.f; }
    {
      size_t krow0 = (size_t)(b * T + kb + l16) * D + h * HD + half * 8;
      bf16x16 bk00 = ld_frag_b(k + krow0);          // dims 0..31
      bf16x16 bk01 = ld_frag_b(k + krow0 + 32);     // dims 32..63
      s0 = wmma_bf16(a0, bk00, s0);
      s0 = wmma_bf16(a1, bk01, s0);
      size_t krow1 = krow0 + (size_t)16 * D;
      bf16x16 bk10 = ld_frag_b(k + krow1);
      bf16x16 bk11 = ld_frag_b(k + krow1 + 32);
      s1 = wmma_bf16(a0, bk10, s1);
      s1 = wmma_bf16(a1, bk11, s1);
    }
    // ---- scale + causal mask + online softmax ----
    float corr[8];
#pragma unroll
    for (int r = 0; r < 8; r++) {
      int mrow = qb + r + 8 * half;
      s0[r] = (kb + l16      <= mrow) ? s0[r] * 0.125f : -__builtin_inff();
      s1[r] = (kb + 16 + l16 <= mrow) ? s1[r] * 0.125f : -__builtin_inff();
      float tm = redmax16(fmaxf(s0[r], s1[r]));
      float nm = fmaxf(rmax[r], tm);
      corr[r] = __expf(rmax[r] - nm);
      rmax[r] = nm;
      float p0 = __expf(s0[r] - nm);
      float p1 = __expf(s1[r] - nm);
      rsum[r] = rsum[r] * corr[r] + redsum16(p0 + p1);
      int row = r + 8 * half;
      pbuf[row * 32 + l16]      = (__bf16)p0;
      pbuf[row * 32 + 16 + l16] = (__bf16)p1;
    }
#pragma unroll
    for (int nf = 0; nf < 4; nf++)
#pragma unroll
      for (int r = 0; r < 8; r++) o[nf][r] *= corr[r];

    // LDS ops are in-order within a wave; make dependence explicit anyway.
    asm volatile("s_wait_dscnt 0" ::: "memory");

    bf16x16 pa = ld_frag_a(pbuf + l16 * 32 + half * 8);  // P as A-operand
    const __bf16* vb = vbase + (size_t)l16 * T + kb + half * 8;
#pragma unroll
    for (int nf = 0; nf < 4; nf++) {
      bf16x16 bv = ld_frag_b(vb + (size_t)nf * 16 * T);
      o[nf] = wmma_bf16(pa, bv, o[nf]);
    }
  }

  // ---- normalize and store concat-head output [B,T,D] bf16 ----
#pragma unroll
  for (int r = 0; r < 8; r++) {
    float inv = 1.0f / rsum[r];
    int row = qb + r + 8 * half;
    size_t base = (size_t)(b * T + row) * D + h * HD + l16;
#pragma unroll
    for (int nf = 0; nf < 4; nf++)
      y[base + nf * 16] = (__bf16)(o[nf][r] * inv);
  }
}

// ---------------- host orchestration ----------------
extern "C" void kernel_launch(void* const* d_in, const int* in_sizes, int n_in,
                              void* d_out, int out_size, void* d_ws, size_t ws_size,
                              hipStream_t stream) {
  const int*   tok  = (const int*)  d_in[0];
  const float* temb = (const float*)d_in[1];
  const float* pemb = (const float*)d_in[2];
  const float* Wq   = (const float*)d_in[3];
  const float* Wk   = (const float*)d_in[4];
  const float* Wv   = (const float*)d_in[5];
  const float* Wo   = (const float*)d_in[6];
  const float* bo   = (const float*)d_in[7];
  const float* ln1g = (const float*)d_in[8];
  const float* ln1b = (const float*)d_in[9];
  const float* ln2g = (const float*)d_in[10];
  const float* ln2b = (const float*)d_in[11];
  const float* W1   = (const float*)d_in[12];
  const float* b1   = (const float*)d_in[13];
  const float* W2   = (const float*)d_in[14];
  const float* b2   = (const float*)d_in[15];
  const float* lnfg = (const float*)d_in[16];
  const float* lnfb = (const float*)d_in[17];
  const float* Wd   = (const float*)d_in[18];
  const float* bd   = (const float*)d_in[19];
  float* out = (float*)d_out;

  char* p = (char*)d_ws;
  auto alloc = [&](size_t bytes) -> char* {
    char* r = p; p += (bytes + 255) & ~(size_t)255; return r;
  };
  __bf16* wqT = (__bf16*)alloc((size_t)L * D * D * 2);
  __bf16* wkT = (__bf16*)alloc((size_t)L * D * D * 2);
  __bf16* wvT = (__bf16*)alloc((size_t)L * D * D * 2);
  __bf16* woT = (__bf16*)alloc((size_t)L * D * D * 2);
  __bf16* w1T = (__bf16*)alloc((size_t)L * D * FF * 2);
  __bf16* w2T = (__bf16*)alloc((size_t)L * FF * D * 2);
  __bf16* wdT = (__bf16*)alloc((size_t)V * D * 2);
  float*  x   = (float*) alloc((size_t)MT * D * 4);
  __bf16* hb  = (__bf16*)alloc((size_t)MT * D * 2);
  __bf16* qb  = (__bf16*)alloc((size_t)MT * D * 2);
  __bf16* kb  = (__bf16*)alloc((size_t)MT * D * 2);
  __bf16* vtb = (__bf16*)alloc((size_t)MT * D * 2);
  __bf16* yb  = (__bf16*)alloc((size_t)MT * D * 2);
  __bf16* fb  = (__bf16*)alloc((size_t)MT * FF * 2);

  dim3 blk256(256);

  // embeddings
  embed_ker<<<MT, blk256, 0, stream>>>(tok, temb, pemb, x);

  // weight transpose + bf16 convert
  for (int l = 0; l < L; l++) {
    transpose_cvt_ker<<<dim3(D / 32, D / 32), blk256, 0, stream>>>(
        Wq + (size_t)l * D * D, wqT + (size_t)l * D * D, D, D);
    transpose_cvt_ker<<<dim3(D / 32, D / 32), blk256, 0, stream>>>(
        Wk + (size_t)l * D * D, wkT + (size_t)l * D * D, D, D);
    transpose_cvt_ker<<<dim3(D / 32, D / 32), blk256, 0, stream>>>(
        Wv + (size_t)l * D * D, wvT + (size_t)l * D * D, D, D);
    transpose_cvt_ker<<<dim3(D / 32, D / 32), blk256, 0, stream>>>(
        Wo + (size_t)l * D * D, woT + (size_t)l * D * D, D, D);
    transpose_cvt_ker<<<dim3(FF / 32, D / 32), blk256, 0, stream>>>(
        W1 + (size_t)l * D * FF, w1T + (size_t)l * D * FF, D, FF);
    transpose_cvt_ker<<<dim3(D / 32, FF / 32), blk256, 0, stream>>>(
        W2 + (size_t)l * FF * D, w2T + (size_t)l * FF * D, FF, D);
  }
  transpose_cvt_ker<<<dim3(V / 32, D / 32), blk256, 0, stream>>>(Wd, wdT, D, V);

  for (int l = 0; l < L; l++) {
    const __bf16* wq_l = wqT + (size_t)l * D * D;
    const __bf16* wk_l = wkT + (size_t)l * D * D;
    const __bf16* wv_l = wvT + (size_t)l * D * D;
    const __bf16* wo_l = woT + (size_t)l * D * D;
    const __bf16* w1_l = w1T + (size_t)l * D * FF;
    const __bf16* w2_l = w2T + (size_t)l * FF * D;

    // h = LN1(x)  (bf16)
    ln_bf16_ker<<<MT, blk256, 0, stream>>>(x, ln1g + (size_t)l * D,
                                           ln1b + (size_t)l * D, hb);
    // Q, K (bf16 [B,T,D]), V (bf16 transposed per head [B,H,HD,T])
    gemm_bf16_ker<MB_BF16><<<dim3(D / 64, MT / 128), blk256, 0, stream>>>(
        hb, wq_l, nullptr, nullptr, nullptr, qb, MT, D, D);
    gemm_bf16_ker<MB_BF16><<<dim3(D / 64, MT / 128), blk256, 0, stream>>>(
        hb, wk_l, nullptr, nullptr, nullptr, kb, MT, D, D);
    gemm_bf16_ker<MB_VT><<<dim3(D / 64, MT / 128), blk256, 0, stream>>>(
        hb, wv_l, nullptr, nullptr, nullptr, vtb, MT, D, D);
    // attention -> yb (bf16 [B,T,D])
    attn_ker<<<B * H * (T / 16), dim3(32), 0, stream>>>(qb, kb, vtb, yb);
    // x = x + yb @ Wo + bo     (in-place residual)
    gemm_bf16_ker<MB_BIAS | MB_RES><<<dim3(D / 64, MT / 128), blk256, 0, stream>>>(
        yb, wo_l, bo + (size_t)l * D, x, x, nullptr, MT, D, D);
    // h2 = LN2(x)
    ln_bf16_ker<<<MT, blk256, 0, stream>>>(x, ln2g + (size_t)l * D,
                                           ln2b + (size_t)l * D, hb);
    // f = relu(h2 @ W1 + b1)   (bf16)
    gemm_bf16_ker<MB_BIAS | MB_RELU | MB_BF16>
        <<<dim3(FF / 64, MT / 128), blk256, 0, stream>>>(
            hb, w1_l, b1 + (size_t)l * FF, nullptr, nullptr, fb, MT, FF, D);
    // x = x + f @ W2 + b2
    gemm_bf16_ker<MB_BIAS | MB_RES><<<dim3(D / 64, MT / 128), blk256, 0, stream>>>(
        fb, w2_l, b2 + (size_t)l * D, x, x, nullptr, MT, D, FF);
  }

  // final LN + logits
  ln_bf16_ker<<<MT, blk256, 0, stream>>>(x, lnfg, lnfb, hb);
  gemm_bf16_ker<MB_BIAS><<<dim3(V / 64, MT / 128), blk256, 0, stream>>>(
      hb, wdT, bd, nullptr, out, nullptr, MT, V, D);
}
